// DynamicDWConv__74371653698147
// MI455X (gfx1250) — matrix operand order repaired and checked by reading, not compiled, not run on hardware
//
#include <hip/hip_runtime.h>
#include <hip/hip_bf16.h>
#include <math.h>

typedef __attribute__((ext_vector_type(16))) _Float16 v16h;
typedef __attribute__((ext_vector_type(8)))  float    v8f;

#define B_   32
#define C_   256
#define H_   64
#define W_   64
#define HW_  (H_ * W_)          // 4096
#define BC_  (B_ * C_)          // 8192
#define N9_  (C_ * 9)           // 2304
#define KDIM 256

// ---------------------------------------------------------------------------
// Kernel 1: global avg + max pool per (b,c) plane.  One block per plane,
// 256 threads (8 wave32s).  float4 loads -> wave shfl reduce -> LDS reduce.
// ---------------------------------------------------------------------------
__global__ void pool_kernel(const float* __restrict__ x,
                            float* __restrict__ avg,
                            float* __restrict__ mx) {
    int bc = blockIdx.x;
    const float4* p4 = (const float4*)(x + (size_t)bc * HW_);
    int tid = threadIdx.x;

    float s = 0.f;
    float m = -INFINITY;
    #pragma unroll
    for (int i = 0; i < 4; ++i) {               // 1024 float4 / 256 threads
        float4 v = p4[tid + i * 256];
        s += (v.x + v.y) + (v.z + v.w);
        m = fmaxf(m, fmaxf(fmaxf(v.x, v.y), fmaxf(v.z, v.w)));
    }
    // wave32 reduction
    #pragma unroll
    for (int off = 16; off > 0; off >>= 1) {
        s += __shfl_xor(s, off, 32);
        m = fmaxf(m, __shfl_xor(m, off, 32));
    }
    __shared__ float ss[8], sm[8];
    int wave = tid >> 5;
    if ((tid & 31) == 0) { ss[wave] = s; sm[wave] = m; }
    __syncthreads();
    if (tid == 0) {
        float S = ss[0], M = sm[0];
        #pragma unroll
        for (int w = 1; w < 8; ++w) { S += ss[w]; M = fmaxf(M, sm[w]); }
        avg[bc] = S * (1.0f / (float)HW_);
        mx[bc]  = M;
    }
}

// ---------------------------------------------------------------------------
// Kernel 2: wt = softmax_{axis of size 1}( (mx@w2^T + b2) - (avg@w1^T + b1) )
// M=32, N=2304, K=256.  One wave (32 threads) per 16x16 output tile.
// Both GEMMs fused: two v8f accumulators, 8 WMMA steps of K=32 each.
// v_wmma_f32_16x16x32_f16; precision irrelevant (softmax over singleton = 1).
// ---------------------------------------------------------------------------
__global__ void gemm_wt_kernel(const float* __restrict__ avg,
                               const float* __restrict__ mx,
                               const float* __restrict__ w1,
                               const float* __restrict__ b1,
                               const float* __restrict__ w2,
                               const float* __restrict__ b2,
                               float* __restrict__ wt) {
    int tile = blockIdx.x;                 // 0 .. 287
    int mt = tile / (N9_ / 16);            // 0..1
    int nt = tile % (N9_ / 16);            // 0..143
    int m0 = mt * 16, n0 = nt * 16;

    int lane = threadIdx.x;                // 0..31 (EXEC all ones for WMMA)
    int lm   = lane & 15;
    bool hi  = lane >= 16;
    int ncol = n0 + lm;                    // output column handled by this lane

    // A fragment K-offsets per ISA layout (16-bit A 16x32):
    //   lanes 0-15 : v0..v3 <- K 0..7,  v4..v7 <- K 16..23
    //   lanes16-31 : v0..v3 <- K 8..15, v4..v7 <- K 24..31
    int akb = hi ? 8 : 0;
    // B fragment (32x16): lane holds column N=lm; same K order as A
    int bkb = hi ? 8 : 0;

    v8f acc1 = {};   // avg @ w1^T
    v8f acc2 = {};   // mx  @ w2^T

    for (int k0 = 0; k0 < KDIM; k0 += 32) {
        v16h a1, a2, bf1, bf2;
        const float* ar1 = avg + (size_t)(m0 + lm) * KDIM + k0 + akb;
        const float* ar2 = mx  + (size_t)(m0 + lm) * KDIM + k0 + akb;
        #pragma unroll
        for (int i = 0; i < 8; ++i) {
            a1[i]     = (_Float16)ar1[i];
            a1[8 + i] = (_Float16)ar1[16 + i];
            a2[i]     = (_Float16)ar2[i];
            a2[8 + i] = (_Float16)ar2[16 + i];
        }
        // out[m,n] = sum_k A[m,k] * W[n,k]  ->  B[k,n] = W[n,k]; contiguous in k
        const float* br1 = w1 + (size_t)ncol * KDIM + k0 + bkb;
        const float* br2 = w2 + (size_t)ncol * KDIM + k0 + bkb;
        #pragma unroll
        for (int i = 0; i < 16; ++i) {
            bf1[i] = (_Float16)br1[i < 8 ? i : i + 8];
            bf2[i] = (_Float16)br2[i < 8 ? i : i + 8];
        }
        acc1 = __builtin_amdgcn_wmma_f32_16x16x32_f16(
                   false, a1, false, bf1, (short)0, acc1, false, false);
        acc2 = __builtin_amdgcn_wmma_f32_16x16x32_f16(
                   false, a2, false, bf2, (short)0, acc2, false, false);
    }

    float bd = b2[ncol] - b1[ncol];
    // D layout: VGPR r -> row m0+r (lanes 0-15) / m0+r+8 (lanes 16-31), col=ncol
    #pragma unroll
    for (int r = 0; r < 8; ++r) {
        int m = m0 + r + (hi ? 8 : 0);
        float t = (acc2[r] - acc1[r]) + bd;
        // softmax over a size-1 axis: exp(t - max) / sum == exp(t - t)
        // (== 1.0 for finite t, NaN propagates for NaN/Inf — faithful)
        wt[(size_t)m * N9_ + ncol] = __expf(t - t);
    }
}

// ---------------------------------------------------------------------------
// Kernel 3: per-sample depthwise 3x3 conv (pad 1) + bias + residual.
// One block per (b,c) plane, 256 threads, 66x66 halo tile in LDS,
// 16 output pixels per thread.
// ---------------------------------------------------------------------------
__global__ void dwconv_kernel(const float* __restrict__ x,
                              const float* __restrict__ wt,
                              const float* __restrict__ bias,
                              float* __restrict__ out) {
    int bc = blockIdx.x;
    int c  = bc & (C_ - 1);
    const float* xp = x   + (size_t)bc * HW_;
    float*       op = out + (size_t)bc * HW_;

    __shared__ float tile[66 * 66];
    int tid = threadIdx.x;

    for (int idx = tid; idx < 66 * 66; idx += 256) {
        int r  = idx / 66, cc = idx - r * 66;
        int gh = r - 1,    gw = cc - 1;
        float v = 0.f;
        if ((unsigned)gh < (unsigned)H_ && (unsigned)gw < (unsigned)W_)
            v = xp[gh * W_ + gw];
        tile[idx] = v;
    }

    float w9[9];
    const float* wp = wt + (size_t)bc * 9;
    #pragma unroll
    for (int j = 0; j < 9; ++j) w9[j] = wp[j];
    float bb = bias[c];

    __syncthreads();

    #pragma unroll
    for (int i = 0; i < 16; ++i) {
        int p = tid + i * 256;
        int h = p >> 6, w = p & 63;
        const float* t0 = &tile[h * 66 + w];   // top-left of 3x3 window
        float acc;
        acc  = fmaf(w9[0], t0[0],   0.f);
        acc  = fmaf(w9[1], t0[1],   acc);
        acc  = fmaf(w9[2], t0[2],   acc);
        acc  = fmaf(w9[3], t0[66],  acc);
        acc  = fmaf(w9[4], t0[67],  acc);
        acc  = fmaf(w9[5], t0[68],  acc);
        acc  = fmaf(w9[6], t0[132], acc);
        acc  = fmaf(w9[7], t0[133], acc);
        acc  = fmaf(w9[8], t0[134], acc);
        op[p] = acc + bb + t0[67];             // + bias + residual x
    }
}

// ---------------------------------------------------------------------------
extern "C" void kernel_launch(void* const* d_in, const int* in_sizes, int n_in,
                              void* d_out, int out_size, void* d_ws, size_t ws_size,
                              hipStream_t stream) {
    const float* x    = (const float*)d_in[0];
    const float* w1   = (const float*)d_in[1];
    const float* b1   = (const float*)d_in[2];
    const float* w2   = (const float*)d_in[3];
    const float* b2   = (const float*)d_in[4];
    const float* bias = (const float*)d_in[5];
    float* out = (float*)d_out;

    float* ws  = (float*)d_ws;
    float* avg = ws;                 // 8192 floats
    float* mx  = ws + BC_;           // 8192 floats
    float* wt  = ws + 2 * BC_;       // 32*2304 = 73728 floats

    pool_kernel<<<BC_, 256, 0, stream>>>(x, avg, mx);
    gemm_wt_kernel<<<(B_ / 16) * (N9_ / 16), 32, 0, stream>>>(avg, mx, w1, b1, w2, b2, wt);
    dwconv_kernel<<<BC_, 256, 0, stream>>>(x, wt, bias, out);
}
